// NeighborhoodAggregator_45681272160925
// MI455X (gfx1250) — matrix operand order repaired and checked by reading, not compiled, not run on hardware
//
#include <hip/hip_runtime.h>
#include <hip/hip_bf16.h>
#include <math.h>
#include <stdint.h>

typedef __attribute__((ext_vector_type(16))) _Float16 v16h;
typedef __attribute__((ext_vector_type(8)))  float    v8f;
typedef __attribute__((ext_vector_type(4)))  float    v4f;
typedef __attribute__((ext_vector_type(2)))  float    v2f;

#define E_DIM 256
#define R_DIM 256
#define HID   512
#define BATCH 1024
#define HLEN  200
#define NCHUNK 13   /* ceil(200/16) */
#define BPB    2    /* batch rows per block (W_rel fragment reuse) */

// ---------------------------------------------------------------------------
// Fused: messages GEMM (f16 WMMA) + time decay + masked flash-softmax attn +
// PNA mean/max aggregation. One block (256 thr = 8 waves) per BPB batch rows.
// Double-buffered LDS staging: ent tile via global_load_async_to_lds_b128
// (ASYNCcnt), rel tile software-pipelined through VGPRs with f32->f16 cvt.
// ---------------------------------------------------------------------------
__global__ __launch_bounds__(256)
void msg_pna_kernel(const float* __restrict__ e_emb,
                    const float* __restrict__ ent,     // (B,H,E)
                    const float* __restrict__ rel,     // (B,H,R)
                    const int*   __restrict__ time_raw,
                    const int*   __restrict__ qtime,
                    const uint8_t* __restrict__ mask,
                    const float* __restrict__ W_rel,   // (E,R)
                    const float* __restrict__ q_repr,  // (B,E)
                    const float* __restrict__ log_gamma,
                    float* __restrict__ combined)      // (B,4E)
{
  const int tid  = threadIdx.x;
  const int lane = tid & 31;
  const int wave = tid >> 5;
  const int g    = (lane >> 4) & 1;   // lane half
  const int nIdx = lane & 15;         // column index within 16-wide tile

  __shared__ _Float16 rel_s[2][16 * 256];   // 2 x 8 KB (f16 A tiles)
  __shared__ float    ent_s[2][16 * 256];   // 2 x 16 KB
  __shared__ float    attn_s[2][16];
  __shared__ float    decay_s[2][16];
  __shared__ float    valid_s[2][16];

  // ---- W_rel fragments resident in VGPRs: B[k][n] = W_rel[e][k] ----------
  // B 32x16 f16 frag in this code's k order (same as A): lane (n = lane&15, g): VGPR v holds K = 16g+2v,+1
  union BF { v16h h; _Float16 e[16]; };
  BF bfrag[2][8];
  const int ebase = wave * 32;
  #pragma unroll
  for (int t = 0; t < 2; ++t) {
    const int e = ebase + t * 16 + nIdx;
    const float* wr = W_rel + (size_t)e * R_DIM;
    #pragma unroll
    for (int kc = 0; kc < 8; ++kc) {
      #pragma unroll
      for (int v = 0; v < 8; ++v) {
        const int k = kc * 32 + g * 16 + 2 * v;
        v2f w = *(const v2f*)(wr + k);
        bfrag[t][kc].e[2 * v]     = (_Float16)w.x;
        bfrag[t][kc].e[2 * v + 1] = (_Float16)w.y;
      }
    }
  }

  const float gamma = __expf(log_gamma[0]);

  for (int bi = 0; bi < BPB; ++bi) {
    const int b = blockIdx.x * BPB + bi;

    const float qt  = (float)qtime[b];
    const float qv0 = q_repr[(size_t)b * E_DIM + ebase + nIdx];
    const float qv1 = q_repr[(size_t)b * E_DIM + ebase + 16 + nIdx];

    float mean0 = 0.f, mean1 = 0.f;
    float max0  = 0.f, max1  = 0.f;   // max over (valid ? msg : 0), as in ref
    float aacc0 = 0.f, aacc1 = 0.f;   // streaming-softmax weighted msg accum
    float m_run = -1e30f, l_run = 0.f;
    float nvald = 0.f;

    auto prefetch_rel = [&](int cc, v4f* dst) {
      #pragma unroll
      for (int i = 0; i < 4; ++i) {
        const int idx = (tid + i * 256) * 4;
        int h = cc * 16 + (idx >> 8); if (h >= HLEN) h = HLEN - 1;
        dst[i] = *(const v4f*)(rel + ((size_t)b * HLEN + h) * 256 + (idx & 255));
      }
    };
    auto issue_ent_async = [&](int cc, int bb) {
      #pragma unroll
      for (int i = 0; i < 4; ++i) {
        const int idx = (tid + i * 256) * 4;
        int h = cc * 16 + (idx >> 8); if (h >= HLEN) h = HLEN - 1;
        const float* ga = ent + ((size_t)b * HLEN + h) * 256 + (idx & 255);
        const uint32_t la = (uint32_t)(uintptr_t)(&ent_s[bb][idx]);
        asm volatile("global_load_async_to_lds_b128 %0, %1, off"
                     :: "v"(la), "v"(ga) : "memory");
      }
    };

    __syncthreads();              // buffers free from previous batch row
    v4f rv[4], rvn[4];
    prefetch_rel(0, rv);
    issue_ent_async(0, 0);

    for (int c = 0; c < NCHUNK; ++c) {
      const int buf = c & 1;
      const int h0  = c * 16;

      // ---- Phase A: commit rel regs -> LDS f16; per-h meta ---------------
      #pragma unroll
      for (int i = 0; i < 4; ++i) {
        const int idx = (tid + i * 256) * 4;
        union { _Float16 h4[4]; uint64_t u; } hp;
        hp.h4[0] = (_Float16)rv[i].x; hp.h4[1] = (_Float16)rv[i].y;
        hp.h4[2] = (_Float16)rv[i].z; hp.h4[3] = (_Float16)rv[i].w;
        *(uint64_t*)(&rel_s[buf][idx]) = hp.u;
      }
      if (tid < 16) {
        const int h = h0 + tid;
        float vld = 0.f, dec = 0.f;
        if (h < HLEN) {
          vld = mask[(size_t)b * HLEN + h] ? 1.f : 0.f;
          float td = fmaxf(qt - (float)time_raw[(size_t)b * HLEN + h], 0.f);
          dec = __expf(-gamma * td);
        }
        valid_s[buf][tid] = vld;
        decay_s[buf][tid] = dec;
        attn_s[buf][tid]  = 0.f;
      }
      asm volatile("s_wait_asynccnt 0x0" ::: "memory");  // ent_s[buf] ready
      __syncthreads();

      // ---- Phase B: prefetch next chunk, then WMMA on current ------------
      if (c + 1 < NCHUNK) {
        issue_ent_async(c + 1, buf ^ 1);
        prefetch_rel(c + 1, rvn);
      }

      v8f c0 = {}; v8f c1 = {};
      union AF { v16h h; uint32_t u[8]; } af;
      const _Float16* rs = rel_s[buf];
      const float*    es = ent_s[buf];
      const int m = nIdx;   // A-matrix row for this lane
      #pragma unroll
      for (int kc = 0; kc < 8; ++kc) {
        #pragma unroll
        for (int v = 0; v < 8; ++v) {
          const int k = kc * 32 + 16 * g + 2 * v;
          af.u[v] = *(const uint32_t*)(rs + m * 256 + k);
        }
        c0 = __builtin_amdgcn_wmma_f32_16x16x32_f16(false, af.h, false,
                bfrag[0][kc].h, (short)0, c0, false, false);
        c1 = __builtin_amdgcn_wmma_f32_16x16x32_f16(false, af.h, false,
                bfrag[1][kc].h, (short)0, c1, false, false);
      }

      // ---- epilogue: msg = C * ent; mean/max/attn-dot --------------------
      float msg0[8], msg1[8], p[8], vl[8];
      #pragma unroll
      for (int v = 0; v < 8; ++v) {
        const int mm = v + 8 * g;             // C row: M = v + 8*laneHalf
        vl[v] = valid_s[buf][mm];
        const float mg0 = c0[v] * es[mm * 256 + ebase + nIdx];
        const float mg1 = c1[v] * es[mm * 256 + ebase + 16 + nIdx];
        msg0[v] = mg0; msg1[v] = mg1;
        mean0 += mg0 * vl[v];  mean1 += mg1 * vl[v];
        max0 = fmaxf(max0, vl[v] != 0.f ? mg0 : 0.f);
        max1 = fmaxf(max1, vl[v] != 0.f ? mg1 : 0.f);
        p[v] = qv0 * mg0 + qv1 * mg1;         // partial q . msg
        nvald += vl[v] * 0.0625f;             // 16 column lanes count once
      }
      #pragma unroll
      for (int s = 1; s < 16; s <<= 1) {
        #pragma unroll
        for (int v = 0; v < 8; ++v) p[v] += __shfl_xor(p[v], s, 32);
      }
      if (nIdx == 0) {
        #pragma unroll
        for (int v = 0; v < 8; ++v) atomicAdd(&attn_s[buf][v + 8 * g], p[v]);
      }
      __syncthreads();

      // ---- Phase C: streaming softmax update -----------------------------
      float aL[8], cmax = -1e30f;
      #pragma unroll
      for (int v = 0; v < 8; ++v) {
        const int mm = v + 8 * g;
        aL[v] = (valid_s[buf][mm] != 0.f)
                  ? attn_s[buf][mm] * decay_s[buf][mm] * 0.0625f  // /sqrt(256)
                  : -1e9f;
        cmax = fmaxf(cmax, aL[v]);
      }
      cmax = fmaxf(cmax, __shfl_xor(cmax, 16, 32));
      const float mnew = fmaxf(m_run, cmax);
      const float corr = __expf(m_run - mnew);
      float wv[8], ls = 0.f;
      #pragma unroll
      for (int v = 0; v < 8; ++v) { wv[v] = __expf(aL[v] - mnew); ls += wv[v]; }
      ls += __shfl_xor(ls, 16, 32);
      l_run = l_run * corr + ls;
      m_run = mnew;
      float s0 = 0.f, s1 = 0.f;
      #pragma unroll
      for (int v = 0; v < 8; ++v) { s0 += wv[v] * msg0[v]; s1 += wv[v] * msg1[v]; }
      aacc0 = aacc0 * corr + s0;
      aacc1 = aacc1 * corr + s1;

      if (c + 1 < NCHUNK) {
        #pragma unroll
        for (int i = 0; i < 4; ++i) rv[i] = rvn[i];
      }
    }

    // ---- combine lane halves and write combined[b] -------------------------
    mean0 += __shfl_xor(mean0, 16, 32);  mean1 += __shfl_xor(mean1, 16, 32);
    aacc0 += __shfl_xor(aacc0, 16, 32);  aacc1 += __shfl_xor(aacc1, 16, 32);
    max0 = fmaxf(max0, __shfl_xor(max0, 16, 32));
    max1 = fmaxf(max1, __shfl_xor(max1, 16, 32));
    nvald += __shfl_xor(nvald, 16, 32);

    const float inv_l = 1.f / l_run;
    const float inv_n = 1.f / fmaxf(nvald, 1.f);
    if (g == 0) {
      const int e0 = ebase + nIdx, e1 = e0 + 16;
      float* out = combined + (size_t)b * 1024;
      out[e0]       = mean0 * inv_n;  out[e1]       = mean1 * inv_n;
      out[256 + e0] = max0;           out[256 + e1] = max1;
      out[512 + e0] = aacc0 * inv_l;  out[512 + e1] = aacc1 * inv_l;
    }
    combined[(size_t)b * 1024 + 768 + tid] = e_emb[(size_t)b * E_DIM + tid];
  }
}

// ---------------------------------------------------------------------------
// f32 GEMM: C[M,N] = A[M,K] @ W[N,K]^T + bias, via v_wmma_f32_16x16x4_f32.
// One 16x16 tile per wave; 8 waves per block. grid = (M/16, N/128).
// ---------------------------------------------------------------------------
__global__ __launch_bounds__(256)
void gemm_f32_kernel(const float* __restrict__ A,
                     const float* __restrict__ W,
                     const float* __restrict__ bias,
                     float* __restrict__ C,
                     int M, int N, int K)
{
  const int wave = threadIdx.x >> 5;
  const int lane = threadIdx.x & 31;
  const int g    = lane >> 4;
  const int nIdx = lane & 15;
  const int row0 = blockIdx.x * 16;
  const int col0 = (blockIdx.y * 8 + wave) * 16;
  if (col0 >= N || row0 >= M) return;

  const float* arow = A + (size_t)(row0 + nIdx) * K;   // A 16x4: lane m=lane&15
  const float* wrow = W + (size_t)(col0 + nIdx) * K;   // B  4x16: lane n=lane&15
  v8f c = {};
  for (int k = 0; k < K; k += 4) {
    v2f a = *(const v2f*)(arow + k + 2 * g);   // V0:K=2g, V1:K=2g+1
    v2f b = *(const v2f*)(wrow + k + 2 * g);
    c = __builtin_amdgcn_wmma_f32_16x16x4_f32(false, a, false, b,
                                              (short)0, c, false, false);
  }
  const float bb = bias ? bias[col0 + nIdx] : 0.f;
  #pragma unroll
  for (int v = 0; v < 8; ++v) {
    C[(size_t)(row0 + v + 8 * g) * N + col0 + nIdx] = c[v] + bb;
  }
}

// ---------------------------------------------------------------------------
// Row LayerNorm (+ optional exact GELU). One block (256 thr) per row, D<=512.
// ---------------------------------------------------------------------------
__global__ __launch_bounds__(256)
void ln_kernel(const float* __restrict__ X, const float* __restrict__ gw,
               const float* __restrict__ bw, float* __restrict__ Y,
               int D, int do_gelu)
{
  const int row = blockIdx.x, tid = threadIdx.x;
  __shared__ float sbuf[256];
  const float* x = X + (size_t)row * D;
  const float x0 = (tid < D) ? x[tid] : 0.f;
  const float x1 = (tid + 256 < D) ? x[tid + 256] : 0.f;

  sbuf[tid] = x0 + x1; __syncthreads();
  for (int o = 128; o > 0; o >>= 1) {
    if (tid < o) sbuf[tid] += sbuf[tid + o];
    __syncthreads();
  }
  const float mu = sbuf[0] / (float)D;
  __syncthreads();
  const float d0 = (tid < D) ? x0 - mu : 0.f;
  const float d1 = (tid + 256 < D) ? x1 - mu : 0.f;
  sbuf[tid] = d0 * d0 + d1 * d1; __syncthreads();
  for (int o = 128; o > 0; o >>= 1) {
    if (tid < o) sbuf[tid] += sbuf[tid + o];
    __syncthreads();
  }
  const float inv = rsqrtf(sbuf[0] / (float)D + 1e-5f);

  if (tid < D) {
    float y = d0 * inv * gw[tid] + bw[tid];
    if (do_gelu) y = 0.5f * y * (1.f + erff(y * 0.70710678118654752f));
    Y[(size_t)row * D + tid] = y;
  }
  if (tid + 256 < D) {
    const int i = tid + 256;
    float y = d1 * inv * gw[i] + bw[i];
    if (do_gelu) y = 0.5f * y * (1.f + erff(y * 0.70710678118654752f));
    Y[(size_t)row * D + i] = y;
  }
}

// ---------------------------------------------------------------------------
extern "C" void kernel_launch(void* const* d_in, const int* in_sizes, int n_in,
                              void* d_out, int out_size, void* d_ws, size_t ws_size,
                              hipStream_t stream) {
  const float*   e_emb    = (const float*)d_in[0];
  const float*   hist_ent = (const float*)d_in[1];
  const float*   hist_rel = (const float*)d_in[2];
  const int*     h_time   = (const int*)d_in[3];
  const int*     q_time   = (const int*)d_in[4];
  const uint8_t* h_mask   = (const uint8_t*)d_in[5];
  const float*   W_rel    = (const float*)d_in[6];
  const float*   W_q      = (const float*)d_in[7];
  const float*   log_g    = (const float*)d_in[8];
  const float*   W1       = (const float*)d_in[9];
  const float*   b1       = (const float*)d_in[10];
  const float*   ln1_g    = (const float*)d_in[11];
  const float*   ln1_b    = (const float*)d_in[12];
  const float*   W2       = (const float*)d_in[13];
  const float*   b2       = (const float*)d_in[14];
  const float*   ln2_g    = (const float*)d_in[15];
  const float*   ln2_b    = (const float*)d_in[16];
  const float*   Wc       = (const float*)d_in[17];
  const float*   bc       = (const float*)d_in[18];
  const float*   lnc_g    = (const float*)d_in[19];
  const float*   lnc_b    = (const float*)d_in[20];

  float* ws      = (float*)d_ws;
  float* q_repr  = ws;                       // 1024*256
  float* comb    = q_repr + BATCH * E_DIM;   // 1024*1024
  float* t1      = comb   + BATCH * 1024;    // 1024*512
  float* h1      = t1     + BATCH * 512;     // 1024*512
  float* t2      = h1     + BATCH * 512;     // 1024*256
  float* t3      = t2     + BATCH * E_DIM;   // 1024*512

  float* dyn = (float*)d_out;                // (B,256)
  float* ctx = dyn + BATCH * E_DIM;          // (B,512)

  // q_repr = e_emb @ W_q^T
  gemm_f32_kernel<<<dim3(BATCH / 16, E_DIM / 128), 256, 0, stream>>>(
      e_emb, W_q, nullptr, q_repr, BATCH, E_DIM, E_DIM);

  // fused messages + decay + flash-softmax attention + PNA -> combined (B,1024)
  msg_pna_kernel<<<dim3(BATCH / BPB), 256, 0, stream>>>(
      e_emb, hist_ent, hist_rel, h_time, q_time, h_mask, W_rel, q_repr,
      log_g, comb);

  // h1 = gelu(LN1(combined @ W1^T + b1))
  gemm_f32_kernel<<<dim3(BATCH / 16, 512 / 128), 256, 0, stream>>>(
      comb, W1, b1, t1, BATCH, 512, 1024);
  ln_kernel<<<dim3(BATCH), 256, 0, stream>>>(t1, ln1_g, ln1_b, h1, 512, 1);

  // dynamic_emb = LN2(h1 @ W2^T + b2)
  gemm_f32_kernel<<<dim3(BATCH / 16, E_DIM / 128), 256, 0, stream>>>(
      h1, W2, b2, t2, BATCH, E_DIM, 512);
  ln_kernel<<<dim3(BATCH), 256, 0, stream>>>(t2, ln2_g, ln2_b, dyn, E_DIM, 0);

  // ctx = gelu(LNc(dynamic_emb @ Wc^T + bc))
  gemm_f32_kernel<<<dim3(BATCH / 16, HID / 128), 256, 0, stream>>>(
      dyn, Wc, bc, t3, BATCH, HID, E_DIM);
  ln_kernel<<<dim3(BATCH), 256, 0, stream>>>(t3, lnc_g, lnc_b, ctx, HID, 1);
}
